// GCNII_Concat_4037269258358
// MI455X (gfx1250) — matrix operand order repaired and checked by reading, not compiled, not run on hardware
//
#include <hip/hip_runtime.h>
#include <hip/hip_bf16.h>
#include <cmath>

typedef __attribute__((ext_vector_type(16))) _Float16 v16h;
typedef __attribute__((ext_vector_type(8)))  float    v8f;

namespace {
constexpr int Fdim = 500;   // input features
constexpr int Hdim = 64;    // hidden
constexpr int Ldim = 64;    // conv layers
constexpr int Rdim = 512;   // mlp out
constexpr int Cdim = 40;    // classes
constexpr int M1   = 213;   // (512-64)/3 + 64
constexpr int M2   = 362;   // 2*(512-64)/3 + 64
// WMMA-friendly padded dims (multiples of 32 for K-loops, 16 for N tiles)
constexpr int FP  = 512;    // 500 -> 512
constexpr int M1P = 224;    // 213 -> 224
constexpr int M2P = 384;    // 362 -> 384
constexpr int CP  = 48;     // 40  -> 48
constexpr float ALPHA = 0.1f;

__host__ __device__ inline int cdiv(int a, int b) { return (a + b - 1) / b; }
}

// ---------------- small utility kernels ----------------

__global__ void gcnii_zero_f32(float* __restrict__ p, int n) {
    int i = blockIdx.x * blockDim.x + threadIdx.x;
    if (i < n) p[i] = 0.f;
}

__global__ void gcnii_deg_scatter(const int* __restrict__ dst, float* __restrict__ deg, int E) {
    int i = blockIdx.x * blockDim.x + threadIdx.x;
    if (i < E) atomicAdd(&deg[dst[i]], 1.0f);
}

// deg -> dinv in place (self-loop adds 1 to every degree, so deg+1 > 0 always)
__global__ void gcnii_dinv(float* __restrict__ deg, int n) {
    int i = blockIdx.x * blockDim.x + threadIdx.x;
    if (i < n) deg[i] = rsqrtf(deg[i] + 1.0f);
}

__global__ void gcnii_norm(const int* __restrict__ src, const int* __restrict__ dst,
                           const float* __restrict__ dinv, float* __restrict__ norm, int E) {
    int i = blockIdx.x * blockDim.x + threadIdx.x;
    if (i < E) norm[i] = dinv[src[i]] * dinv[dst[i]];
}

// Swizzle f32 weight matrices [nmat x rows x cols] (row-major, K x N per matrix) into
// the exact WMMA B-fragment order, f16, zero-padded to [Kpad x Nout]:
//   out[(((mat*ntiles + nt)*ksteps + ks)*32 + lane)*16 + e]
//     = W[mat][ ks*32 + (lane&16 ? 8:0) + e + (e&8 ? 8:0) ][ nt*16 + (lane&15) ]
// so the GEMM can load a whole v16h fragment as one contiguous 32-byte access.
__global__ void gcnii_swizzle_wf(const float* __restrict__ W, int rows, int cols,
                                 _Float16* __restrict__ Wf, int Kpad, int Nout, int nmat) {
    int ksteps = Kpad >> 5, ntiles = Nout >> 4;
    long long tot = (long long)nmat * ntiles * ksteps * 512;
    long long i = (long long)blockIdx.x * blockDim.x + threadIdx.x;
    if (i >= tot) return;
    int e    = (int)(i & 15);
    int lane = (int)((i >> 4) & 31);
    long long rest = i >> 9;
    int ks  = (int)(rest % ksteps);  rest /= ksteps;
    int nt  = (int)(rest % ntiles);
    int mat = (int)(rest / ntiles);
    int k = ks * 32 + ((lane & 16) ? 8 : 0) + e + ((e & 8) ? 8 : 0);
    int n = nt * 16 + (lane & 15);
    float v = (k < rows && n < cols) ? W[((size_t)mat * rows + k) * cols + n] : 0.f;
    Wf[i] = (_Float16)v;
}

// Fused tail weights: Wc = w2 @ out_w  (f32 temp, padded [M2P x CP], swizzled later)
__global__ void gcnii_combine_w(const float* __restrict__ w2, const float* __restrict__ outw,
                                float* __restrict__ wc) {
    int i = blockIdx.x * blockDim.x + threadIdx.x;
    if (i >= M2P * CP) return;
    int k = i / CP, j = i % CP;
    float s = 0.f;
    if (k < M2 && j < Cdim) {
        for (int r = 0; r < Rdim; ++r) s += w2[(size_t)k * Rdim + r] * outw[(size_t)r * Cdim + j];
    }
    wc[i] = s;
}

// bc = b2 @ out_w + out_b  (f32, padded [CP])
__global__ void gcnii_combine_b(const float* __restrict__ b2, const float* __restrict__ outw,
                                const float* __restrict__ outb, float* __restrict__ bc) {
    int j = blockIdx.x * blockDim.x + threadIdx.x;
    if (j >= CP) return;
    float s = 0.f;
    if (j < Cdim) {
        s = outb[j];
        for (int r = 0; r < Rdim; ++r) s += b2[r] * outw[(size_t)r * Cdim + j];
    }
    bc[j] = s;
}

// agg[n,f] = dinv[n]^2 * x[n,f]  (self-loop contribution; also zero-inits accumulator)
__global__ void gcnii_agg_init(const float* __restrict__ x, const float* __restrict__ dinv,
                               float* __restrict__ agg, int n64) {
    int i = blockIdx.x * blockDim.x + threadIdx.x;
    if (i < n64) {
        float d = dinv[i >> 6];
        agg[i] = d * d * x[i];
    }
}

// agg[dst] += norm[e] * x[src]  — one thread per (edge, feature)
__global__ void gcnii_edge_scatter(const int* __restrict__ src, const int* __restrict__ dst,
                                   const float* __restrict__ norm, const float* __restrict__ x,
                                   float* __restrict__ agg, long long total) {
    long long i = (long long)blockIdx.x * blockDim.x + threadIdx.x;
    if (i >= total) return;
    int e = (int)(i >> 6);
    int f = (int)(i & 63);
    atomicAdd(&agg[(size_t)dst[e] * 64 + f], norm[e] * x[(size_t)src[e] * 64 + f]);
}

// ---------------- WMMA GEMM kernels ----------------
// One wave computes a 16 x (NTG*16) output block: A fragment built once per K-step,
// then NTG back-to-back WMMAs into NTG accumulators. 8 waves / 256-thread block.
// Fragment layouts per CDNA5 ISA 7.12.2:
//   A (f16 16x32): lane m=lane%16; lanes 0-15 hold K {0..7,16..23}, lanes 16-31 {8..15,24..31}
//     -> per lane: two runs of 8 consecutive K values => 4x float4 loads from f32 source.
//   B: pre-swizzled to fragment order, loaded as one contiguous v16h (2x b128).
//   C/D (f32 16x16): VGPR r -> M = r + (lane<16 ? 0 : 8), N = lane%16
// M (row count) is assumed to be a multiple of 16 (true here: N=100000).

__device__ __forceinline__ void gcnii_pack_a(v16h& a, float4 q0, float4 q1,
                                             float4 q2, float4 q3) {
    a[0]  = (_Float16)q0.x; a[1]  = (_Float16)q0.y; a[2]  = (_Float16)q0.z; a[3]  = (_Float16)q0.w;
    a[4]  = (_Float16)q1.x; a[5]  = (_Float16)q1.y; a[6]  = (_Float16)q1.z; a[7]  = (_Float16)q1.w;
    a[8]  = (_Float16)q2.x; a[9]  = (_Float16)q2.y; a[10] = (_Float16)q2.z; a[11] = (_Float16)q2.w;
    a[12] = (_Float16)q3.x; a[13] = (_Float16)q3.y; a[14] = (_Float16)q3.z; a[15] = (_Float16)q3.w;
}

// out[m,n] = act( sum_k A[m,k] * W[k,n] + bias[n] ), A f32 row-major (stride lda),
// valid k < Kused (else 0); Wf pre-swizzled f16 fragments for [Kpad x Nout].
// Requires (Nout/16) % NTG == 0.
template <int NTG>
__global__ void gcnii_gemm_wmma(const float* __restrict__ A, int lda, int Kused, int Kpad,
                                const _Float16* __restrict__ Wf, int Nout,
                                const float* __restrict__ bias, int blen,
                                float* __restrict__ out, int ostride, int ocols,
                                int Mtiles, int relu) {
    int ntiles  = Nout >> 4;
    int ksteps  = Kpad >> 5;
    int ngroups = ntiles / NTG;
    int wave = threadIdx.x >> 5;
    int tile = blockIdx.x * (blockDim.x >> 5) + wave;
    if (tile >= Mtiles * ngroups) return;   // wave-uniform: EXEC stays all-ones for WMMA
    int mt  = tile / ngroups;
    int nt0 = (tile % ngroups) * NTG;

    int lane = threadIdx.x & 31;
    int lm   = lane & 15;
    int half = lane >> 4;
    int akb  = half * 8;

    const float* arow = A + (size_t)(mt * 16 + lm) * lda;
    const v16h*  Bf   = (const v16h*)Wf;

    v8f zero = {0.f, 0.f, 0.f, 0.f, 0.f, 0.f, 0.f, 0.f};
    v8f acc[NTG];
#pragma unroll
    for (int t = 0; t < NTG; ++t) acc[t] = zero;

    int ks = 0;
    for (int k0 = 0; k0 + 32 <= Kused; k0 += 32, ++ks) {
        const float4* ap = (const float4*)(arow + k0 + akb);
        v16h a;
        gcnii_pack_a(a, ap[0], ap[1], ap[4], ap[5]);
#pragma unroll
        for (int t = 0; t < NTG; ++t) {
            v16h b = Bf[((size_t)(nt0 + t) * ksteps + ks) * 32 + lane];
            acc[t] = __builtin_amdgcn_wmma_f32_16x16x32_f16(false, a, false, b, (short)0,
                                                            acc[t], false, false);
        }
    }
    if (ks < ksteps) {                      // uniform tail (only when Kused % 32 != 0)
        int k0 = ks * 32;
        v16h a;
#pragma unroll
        for (int e = 0; e < 16; ++e) {
            int ka = k0 + akb + e + ((e & 8) ? 8 : 0);
            int kc = (ka < Kused) ? ka : (Kused - 1);     // clamp address, select value
            float av = arow[kc];
            a[e] = (_Float16)((ka < Kused) ? av : 0.f);
        }
#pragma unroll
        for (int t = 0; t < NTG; ++t) {
            v16h b = Bf[((size_t)(nt0 + t) * ksteps + ks) * 32 + lane];
            acc[t] = __builtin_amdgcn_wmma_f32_16x16x32_f16(false, a, false, b, (short)0,
                                                            acc[t], false, false);
        }
    }

    int rbase = mt * 16 + half * 8;
#pragma unroll
    for (int t = 0; t < NTG; ++t) {
        int n = (nt0 + t) * 16 + lm;
        float bn = (n < blen) ? bias[n] : 0.f;
        if (n < ocols) {
#pragma unroll
            for (int r = 0; r < 8; ++r) {
                float v = acc[t][r] + bn;
                if (relu) v = fmaxf(v, 0.f);
                out[(size_t)(rbase + r) * ostride + n] = v;
            }
        }
    }
}

// GCNII layer: h = 0.9*agg + 0.1*x0 ; out = relu((1-beta)*h + beta*(h @ W)), all [*,64].
// One wave per 16-row block: 4 accumulators cover the full 64-wide output.
__global__ void gcnii_layer_gemm(const float* __restrict__ agg, const float* __restrict__ x0,
                                 const _Float16* __restrict__ Wf, float beta,
                                 float* __restrict__ out, int Mtiles) {
    int wave = threadIdx.x >> 5;
    int mt = blockIdx.x * (blockDim.x >> 5) + wave;
    if (mt >= Mtiles) return;

    int lane = threadIdx.x & 31;
    int lm   = lane & 15;
    int half = lane >> 4;
    int akb  = half * 8;
    size_t arow = (size_t)(mt * 16 + lm) * 64;

    const v16h* Bf = (const v16h*)Wf;

    v8f zero = {0.f, 0.f, 0.f, 0.f, 0.f, 0.f, 0.f, 0.f};
    v8f acc[4];
#pragma unroll
    for (int t = 0; t < 4; ++t) acc[t] = zero;

#pragma unroll
    for (int ks = 0; ks < 2; ++ks) {
        const float4* ap = (const float4*)(agg + arow + ks * 32 + akb);
        const float4* xp = (const float4*)(x0  + arow + ks * 32 + akb);
        float4 q[4];
#pragma unroll
        for (int j = 0; j < 4; ++j) {
            float4 ag = ap[(j & 1) + (j >> 1) * 4];
            float4 x4 = xp[(j & 1) + (j >> 1) * 4];
            q[j].x = (1.f - ALPHA) * ag.x + ALPHA * x4.x;
            q[j].y = (1.f - ALPHA) * ag.y + ALPHA * x4.y;
            q[j].z = (1.f - ALPHA) * ag.z + ALPHA * x4.z;
            q[j].w = (1.f - ALPHA) * ag.w + ALPHA * x4.w;
        }
        v16h a;
        gcnii_pack_a(a, q[0], q[1], q[2], q[3]);
#pragma unroll
        for (int t = 0; t < 4; ++t) {
            v16h b = Bf[((size_t)t * 2 + ks) * 32 + lane];
            acc[t] = __builtin_amdgcn_wmma_f32_16x16x32_f16(false, a, false, b, (short)0,
                                                            acc[t], false, false);
        }
    }

    int rbase = mt * 16 + half * 8;
#pragma unroll
    for (int t = 0; t < 4; ++t) {
        int n = t * 16 + lm;
#pragma unroll
        for (int r = 0; r < 8; ++r) {
            size_t idx = (size_t)(rbase + r) * 64 + n;
            float h = (1.f - ALPHA) * agg[idx] + ALPHA * x0[idx];
            float v = (1.f - beta) * h + beta * acc[t][r];
            out[idx] = fmaxf(v, 0.f);
        }
    }
}

// ---------------- host orchestration ----------------

extern "C" void kernel_launch(void* const* d_in, const int* in_sizes, int n_in,
                              void* d_out, int out_size, void* d_ws, size_t ws_size,
                              hipStream_t stream) {
    (void)n_in; (void)out_size; (void)ws_size;

    const int*   ei      = (const int*)d_in[0];
    const float* x_param = (const float*)d_in[1];
    const float* lin0_w  = (const float*)d_in[2];
    const float* lin0_b  = (const float*)d_in[3];
    const float* conv_w  = (const float*)d_in[4];
    const float* mw0     = (const float*)d_in[5];
    const float* mb0     = (const float*)d_in[6];
    const float* mw1     = (const float*)d_in[7];
    const float* mb1     = (const float*)d_in[8];
    const float* mw2     = (const float*)d_in[9];
    const float* mb2     = (const float*)d_in[10];
    const float* outw    = (const float*)d_in[11];
    const float* outb    = (const float*)d_in[12];

    const int E = in_sizes[0] / 2;
    const int N = in_sizes[1] / Fdim;
    const int* src = ei;
    const int* dst = ei + E;
    const int Mtiles = cdiv(N, 16);

    // workspace layout (256B aligned slices)
    char* wsp = (char*)d_ws;
    auto alloc = [&](size_t bytes) -> char* {
        char* p = wsp;
        wsp += (bytes + 255) & ~(size_t)255;
        return p;
    };
    float*    dinv = (float*)alloc((size_t)N * 4);              // deg -> dinv
    float*    nrm  = (float*)alloc((size_t)E * 4);
    float*    x0   = (float*)alloc((size_t)N * Hdim * 4);
    float*    xA   = (float*)alloc((size_t)N * Hdim * 4);
    float*    xB   = (float*)alloc((size_t)N * Hdim * 4);
    float*    agg  = (float*)alloc((size_t)N * Hdim * 4);
    float*    h1   = (float*)alloc((size_t)N * M1P * 4);
    float*    h2   = (float*)alloc((size_t)N * M2P * 4);
    _Float16* w0f  = (_Float16*)alloc((size_t)FP * Hdim * 2);
    _Float16* cwf  = (_Float16*)alloc((size_t)Ldim * Hdim * Hdim * 2);
    _Float16* m0f  = (_Float16*)alloc((size_t)Hdim * M1P * 2);
    _Float16* m1f  = (_Float16*)alloc((size_t)M1P * M2P * 2);
    _Float16* wcf  = (_Float16*)alloc((size_t)M2P * CP * 2);
    float*    wc32 = (float*)alloc((size_t)M2P * CP * 4);
    float*    bch  = (float*)alloc((size_t)CP * 4);

    const int TB = 256;
    const int WPB = TB / 32;   // waves (tiles) per block

    // ---- graph normalization ----
    gcnii_zero_f32<<<cdiv(N, TB), TB, 0, stream>>>(dinv, N);
    gcnii_deg_scatter<<<cdiv(E, TB), TB, 0, stream>>>(dst, dinv, E);
    gcnii_dinv<<<cdiv(N, TB), TB, 0, stream>>>(dinv, N);
    gcnii_norm<<<cdiv(E, TB), TB, 0, stream>>>(src, dst, dinv, nrm, E);

    // ---- weight conversion: pad + swizzle into WMMA B-fragment order ----
    gcnii_swizzle_wf<<<cdiv(FP * Hdim, TB), TB, 0, stream>>>(lin0_w, Fdim, Hdim, w0f, FP, Hdim, 1);
    gcnii_swizzle_wf<<<cdiv(Ldim * Hdim * Hdim, TB), TB, 0, stream>>>(
        conv_w, Hdim, Hdim, cwf, Hdim, Hdim, Ldim);
    gcnii_swizzle_wf<<<cdiv(Hdim * M1P, TB), TB, 0, stream>>>(mw0, Hdim, M1, m0f, Hdim, M1P, 1);
    gcnii_swizzle_wf<<<cdiv(M1P * M2P, TB), TB, 0, stream>>>(mw1, M1, M2, m1f, M1P, M2P, 1);
    gcnii_combine_w<<<cdiv(M2P * CP, TB), TB, 0, stream>>>(mw2, outw, wc32);
    gcnii_combine_b<<<1, 64, 0, stream>>>(mb2, outw, outb, bch);
    gcnii_swizzle_wf<<<cdiv(M2P * CP, TB), TB, 0, stream>>>(wc32, M2P, CP, wcf, M2P, CP, 1);

    // ---- x0 = relu(x_param @ lin0_w + b) ----  (4 n-tiles, 1 group of NTG=4)
    gcnii_gemm_wmma<4><<<cdiv(Mtiles * 1, WPB), TB, 0, stream>>>(
        x_param, Fdim, Fdim, FP, w0f, Hdim, lin0_b, Hdim,
        x0, Hdim, Hdim, Mtiles, 1);
    hipMemcpyAsync(xA, x0, (size_t)N * Hdim * 4, hipMemcpyDeviceToDevice, stream);

    // ---- 64 GCNII layers ----
    float* xcur = xA;
    float* xnext = xB;
    const long long etot = (long long)E * 64;
    for (int l = 0; l < Ldim; ++l) {
        float beta = logf(0.5f / (float)(l + 1) + 1.0f);
        gcnii_agg_init<<<cdiv(N * Hdim, TB), TB, 0, stream>>>(xcur, dinv, agg, N * Hdim);
        gcnii_edge_scatter<<<(int)((etot + TB - 1) / TB), TB, 0, stream>>>(
            src, dst, nrm, xcur, agg, etot);
        gcnii_layer_gemm<<<cdiv(Mtiles, WPB), TB, 0, stream>>>(
            agg, x0, cwf + (size_t)l * Hdim * Hdim, beta, xnext, Mtiles);
        float* t = xcur; xcur = xnext; xnext = t;
    }

    // ---- MLP (last two layers fused into one GEMM) ----
    // h1 = relu(x @ w0 + b0): Nout=224 -> 14 n-tiles = 2 groups of NTG=7
    gcnii_gemm_wmma<7><<<cdiv(Mtiles * 2, WPB), TB, 0, stream>>>(
        xcur, Hdim, Hdim, Hdim, m0f, M1P, mb0, M1,
        h1, M1P, M1P, Mtiles, 1);
    // h2 = relu(h1 @ w1 + b1): Nout=384 -> 24 n-tiles = 6 groups of NTG=4
    gcnii_gemm_wmma<4><<<cdiv(Mtiles * 6, WPB), TB, 0, stream>>>(
        h1, M1P, M1P, M1P, m1f, M2P, mb1, M2,
        h2, M2P, M2P, Mtiles, 1);
    // out = h2 @ (w2@out_w) + (b2@out_w + out_b): Nout=48 -> 3 n-tiles = 1 group of NTG=3
    gcnii_gemm_wmma<3><<<cdiv(Mtiles * 1, WPB), TB, 0, stream>>>(
        h2, M2P, M2P, M2P, wcf, CP, bch, CP,
        (float*)d_out, Cdim, Cdim, Mtiles, 0);
}
